// AudioRNN_23313082482924
// MI455X (gfx1250) — matrix operand order repaired and checked by reading, not compiled, not run on hardware
//
#include <hip/hip_runtime.h>
#include <hip/hip_bf16.h>

// ---------------------------------------------------------------------------
// LSTM persistent kernel for MI455X (gfx1250, wave32, WMMA), round 3.
// Two persistent workgroups (batch-split, zero inter-WG sync), 16 waves each.
// Wh (bf16, 512KB) partitioned across on-chip stores:
//   K-tiles 0..1 -> VGPRs (64 regs/wave), K-tiles 2..5 -> LDS (256KB),
//   K-tiles 6..7 -> streamed from L2 (loop-invariant addresses, stay hot).
// Gate-local N mapping: wave w owns gate columns {g*256 + w*16 + arow}, so
// i/f/g/o for an h-unit land in one lane's accumulators -> no z LDS buffer.
// Round 3: unified software pipeline with BOTH A and B fragments prefetched
// one K-stage ahead, so per-stage s_wait_dscnt no longer drains to zero.
// ---------------------------------------------------------------------------

typedef __bf16 bf16_t;
typedef bf16_t v16bf __attribute__((ext_vector_type(16)));
typedef bf16_t v8bf  __attribute__((ext_vector_type(8)));
typedef float  v8f   __attribute__((ext_vector_type(8)));

#define T_STEPS 8192
#define Hdim    256
#define G4H     1024
#define WGB     16         // batches per workgroup
#define HS      264        // h row stride (bf16) = 132 dwords -> 4-bank skew
#define KT_REG  2          // K-tiles 0..1 in registers
#define KT_LDS  6          // K-tiles 2..5 in LDS
#define LDS_B_ELEMS (4 * 64 * 512)   // 4 kt-slices * 64 nt * 512 elems = 256KB

// --- fast transcendentals (v_exp_f32 + v_rcp_f32) --------------------------
__device__ __forceinline__ float fsig(float v) {
    return __builtin_amdgcn_rcpf(1.0f + __expf(-v));
}
__device__ __forceinline__ float ftanh(float v) {
    float e = __expf(-2.0f * v);
    return (1.0f - e) * __builtin_amdgcn_rcpf(1.0f + e);
}

// ---------------------------------------------------------------------------
// Pre-swizzle Wh [256,1024] f32 -> bf16 WMMA B-fragments, kt-major:
// frag(kt,nt): lane l holds N col nt*16+l%16, 16 K-rows in table order.
// Element index: ((kt*64 + nt)*32 + l)*16 + s .  kt-slice = 64KB contiguous.
// ---------------------------------------------------------------------------
__global__ void wh_swizzle_kernel(const float* __restrict__ Wh,
                                  bf16_t* __restrict__ dst) {
    int idx = blockIdx.x * blockDim.x + threadIdx.x;
    if (idx >= Hdim * G4H) return;
    int r  = idx >> 10;       // K row 0..255
    int c  = idx & 1023;      // N col 0..1023
    int kt = r >> 5, kk = r & 31;
    int nt = c >> 4, n = c & 15;
    int l = n + (((kk >> 3) & 1) << 4);
    int m = (kk & 7) + ((kk >> 4) << 3);
    dst[(((kt * 64 + nt) * 32 + l) << 4) + m] = (bf16_t)Wh[idx];
}

// ---------------------------------------------------------------------------
__global__ __launch_bounds__(512, 1) void lstm_persistent(
    const float* __restrict__ x,   const float* __restrict__ h0,
    const float* __restrict__ c0,  const float* __restrict__ Wx,
    const float* __restrict__ bgt, const float* __restrict__ Wo,
    const float* __restrict__ bo,  const bf16_t* WhSw,   // no restrict: keep streamed loads in-loop
    float* __restrict__ out_hT, float* __restrict__ out_cT,
    float* __restrict__ out_y)
{
    __shared__ bf16_t blds[LDS_B_ELEMS];  // 256KB: Wh K-tiles 2..5
    __shared__ bf16_t hbuf[WGB * HS];     // 8.25KB hidden state (bf16)
    __shared__ float  pbuf[16 * 16];      // [wave][batch] output partials
    __shared__ float  xs[WGB];            // x_t per batch

    const int tid  = threadIdx.x;
    const int lane = tid & 31;
    const int w    = tid >> 5;            // wave id 0..15
    const int bg0  = blockIdx.x * WGB;    // global batch base
    const int hi   = lane >> 4;           // 0: rows 0..7, 1: rows 8..15
    const int arow = lane & 15;           // M row / h-unit within wave slice
    const int j    = w * 16 + arow;       // this lane's h-unit (column)

    // ---- prologue ---------------------------------------------------------
    {   // h0 -> hbuf (bf16); wave w fills row w
        const int k = lane * 8;
        #pragma unroll
        for (int q = 0; q < 8; ++q)
            hbuf[w * HS + k + q] = (bf16_t)h0[(bg0 + w) * Hdim + k + q];
    }
    // c register-resident: lane owns 8 batches x 1 h-unit
    float creg[8];
    #pragma unroll
    for (int r = 0; r < 8; ++r)
        creg[r] = c0[(bg0 + r + 8 * hi) * Hdim + j];

    const float wo_j = Wo[j];
    const float bo_s = bo[0];

    float wxv[4], bv[4];                  // gate g column = g*256 + j
    #pragma unroll
    for (int g = 0; g < 4; ++g) { wxv[g] = Wx[g * 256 + j]; bv[g] = bgt[g * 256 + j]; }

    if (tid < WGB) xs[tid] = x[(bg0 + tid) * T_STEPS + 0];

    // copy Wh K-tiles 2..5 (256KB contiguous slice) global -> LDS
    {
        const v8bf* src = (const v8bf*)(WhSw + KT_REG * 64 * 512);
        v8bf* dst = (v8bf*)blds;
        for (int cch = tid; cch < LDS_B_ELEMS / 8; cch += 512)
            dst[cch] = src[cch];
    }

    // register-resident B fragments: K-tiles 0..1, all 4 gates
    v16bf breg[KT_REG][4];
    #pragma unroll
    for (int kt = 0; kt < KT_REG; ++kt)
        #pragma unroll
        for (int g = 0; g < 4; ++g)
            breg[kt][g] = *(const v16bf*)(
                WhSw + (((kt * 64 + g * 16 + w) * 32 + lane) << 4));

    __syncthreads();

    for (int t = 0; t < T_STEPS; ++t) {
        // ================= Phase A: z = x_t*Wx + b + h @ Wh ================
        float xr[8];
        #pragma unroll
        for (int r = 0; r < 8; ++r) xr[r] = xs[r + 8 * hi];
        const float xval = xs[w];

        v8f acc[4];
        #pragma unroll
        for (int g = 0; g < 4; ++g)
            #pragma unroll
            for (int r = 0; r < 8; ++r)
                acc[g][r] = xr[r] * wxv[g] + bv[g];

        // --- unified software pipeline over K-tiles 0..7 -------------------
        // A (from hbuf LDS) and B (regs / LDS / L2) both one stage ahead.
        v16bf acur, anxt;
        v16bf bcur[4], bnxt[4];
        {   // stage 0 A fragment
            const bf16_t* ap = &hbuf[arow * HS + 0 * 32 + 8 * hi];
            v8bf alo = *(const v8bf*)(ap);
            v8bf ahi = *(const v8bf*)(ap + 16);
            acur = __builtin_shufflevector(alo, ahi,
                0,1,2,3,4,5,6,7,8,9,10,11,12,13,14,15);
        }
        #pragma unroll
        for (int kt = 0; kt < 8; ++kt) {
            if (kt < 7) {
                const int kn = kt + 1;
                {   // prefetch A(kt+1)
                    const bf16_t* ap = &hbuf[arow * HS + kn * 32 + 8 * hi];
                    v8bf alo = *(const v8bf*)(ap);
                    v8bf ahi = *(const v8bf*)(ap + 16);
                    anxt = __builtin_shufflevector(alo, ahi,
                        0,1,2,3,4,5,6,7,8,9,10,11,12,13,14,15);
                }
                if (kn >= KT_REG) {          // prefetch B(kt+1)
                    if (kn < KT_LDS) {
                        #pragma unroll
                        for (int g = 0; g < 4; ++g)
                            bnxt[g] = *(const v16bf*)(
                                &blds[(((kn - KT_REG) * 64 + g * 16 + w) * 32 + lane) << 4]);
                    } else {
                        #pragma unroll
                        for (int g = 0; g < 4; ++g)
                            bnxt[g] = *(const v16bf*)(
                                WhSw + (((kn * 64 + g * 16 + w) * 32 + lane) << 4));
                    }
                }
            }
            if (kt < KT_REG) {
                #pragma unroll
                for (int g = 0; g < 4; ++g)
                    acc[g] = __builtin_amdgcn_wmma_f32_16x16x32_bf16(
                        false, acur, false, breg[kt][g], (short)0, acc[g], false, false);
            } else {
                #pragma unroll
                for (int g = 0; g < 4; ++g)
                    acc[g] = __builtin_amdgcn_wmma_f32_16x16x32_bf16(
                        false, acur, false, bcur[g], (short)0, acc[g], false, false);
            }
            if (kt < 7) {
                acur = anxt;
                if (kt + 1 >= KT_REG) {
                    #pragma unroll
                    for (int g = 0; g < 4; ++g) bcur[g] = bnxt[g];
                }
            }
        }

        __syncthreads();   // everyone done reading hbuf / xs

        // ============ Phase B: gates straight from accumulators ============
        // acc[g][r] = z_gate(batch m = r+8*hi, unit j); gate order i,f,g,o
        float part[8];
        #pragma unroll
        for (int r = 0; r < 8; ++r) {
            const float cn = fsig(acc[1][r]) * creg[r]
                           + fsig(acc[0][r]) * ftanh(acc[2][r]);
            creg[r] = cn;
            const float hn = fsig(acc[3][r]) * ftanh(cn);
            hbuf[(r + 8 * hi) * HS + j] = (bf16_t)hn;
            part[r] = hn * wo_j;
        }
        // reduce over the 16 arow lanes in each half-wave
        #pragma unroll
        for (int off = 1; off < 16; off <<= 1)
            #pragma unroll
            for (int r = 0; r < 8; ++r)
                part[r] += __shfl_xor(part[r], off, 32);
        if (arow == 0) {
            #pragma unroll
            for (int r = 0; r < 8; ++r)
                pbuf[w * 16 + 8 * hi + r] = part[r];
        }
        __syncthreads();

        // ============ Phase C: output projection, wave w = batch w =========
        {
            float v = (lane < 16) ? pbuf[lane * 16 + w] : 0.0f;
            #pragma unroll
            for (int off = 1; off < 16; off <<= 1)
                v += __shfl_xor(v, off, 32);
            if (lane == 0)
                out_y[(bg0 + w) * T_STEPS + t] = v + bo_s + xval;
        }
        if (w == 0 && lane < WGB && (t + 1) < T_STEPS)
            xs[lane] = x[(bg0 + lane) * T_STEPS + t + 1];

        __syncthreads();
    }

    // ---- epilogue: hT, cT -------------------------------------------------
    {
        const int k = lane * 8;
        #pragma unroll
        for (int q = 0; q < 8; ++q)
            out_hT[(bg0 + w) * Hdim + k + q] = (float)hbuf[w * HS + k + q];
    }
    #pragma unroll
    for (int r = 0; r < 8; ++r)
        out_cT[(bg0 + r + 8 * hi) * Hdim + j] = creg[r];
}

// ---------------------------------------------------------------------------
extern "C" void kernel_launch(void* const* d_in, const int* in_sizes, int n_in,
                              void* d_out, int out_size, void* d_ws, size_t ws_size,
                              hipStream_t stream) {
    const float* x  = (const float*)d_in[0];
    const float* h0 = (const float*)d_in[1];
    const float* c0 = (const float*)d_in[2];
    const float* Wx = (const float*)d_in[3];
    const float* Wh = (const float*)d_in[4];
    const float* b  = (const float*)d_in[5];
    const float* Wo = (const float*)d_in[6];
    const float* bo = (const float*)d_in[7];

    float* out = (float*)d_out;
    float* out_hT = out;                       // [32,256]
    float* out_cT = out + 32 * 256;            // [32,256]
    float* out_y  = out + 2 * 32 * 256;        // [32,8192,1]

    bf16_t* whsw = (bf16_t*)d_ws;              // 512 KB swizzled Wh

    const int nelem = Hdim * G4H;
    wh_swizzle_kernel<<<(nelem + 255) / 256, 256, 0, stream>>>(Wh, whsw);
    lstm_persistent<<<2, 512, 0, stream>>>(x, h0, c0, Wx, b, Wo, bo, whsw,
                                           out_hT, out_cT, out_y);
}
